// MultiHeadSelfAttention_7962869367129
// MI455X (gfx1250) — matrix operand order repaired and checked by reading, not compiled, not run on hardware
//
#include <hip/hip_runtime.h>
#include <hip/hip_bf16.h>

typedef __attribute__((ext_vector_type(16))) _Float16 v16h;
typedef __attribute__((ext_vector_type(8)))  _Float16 v8h;
typedef __attribute__((ext_vector_type(8)))  float    v8f;
typedef __attribute__((ext_vector_type(4)))  unsigned int v4u;
typedef __attribute__((ext_vector_type(8)))  int          v8i;
typedef __attribute__((ext_vector_type(4)))  int          v4i;

__device__ __forceinline__ v8f wmma_f16(v16h a, v16h b, v8f c) {
  // D = A(16x32 f16) * B(32x16 f16) + C(16x16 f32)
  return __builtin_amdgcn_wmma_f32_16x16x32_f16(
      /*neg_a=*/false, a, /*neg_b=*/false, b,
      /*c_mod=*/(short)0, c, /*reuse_a=*/false, /*reuse_b=*/false);
}

__device__ __forceinline__ v8f vzero8f() {
  v8f z;
#pragma unroll
  for (int i = 0; i < 8; ++i) z[i] = 0.0f;
  return z;
}

// ---- TDM: 2D tile load (global -> LDS), element size 2 bytes -------------
// D# per CDNA5 ISA ch.8: group0 = {flags, lds_addr, global_addr, type},
// group1 = {mask/data_size, tensor dims, tile dims, dim0 stride}, groups 2/3
// zero for 2D tiles (tile_dim2 == 0 -> unused).
__device__ __forceinline__ void tdm_load_2d_f16(const void* gptr,
                                                const void* lptr,
                                                unsigned tensor_d0,
                                                unsigned tensor_d1,
                                                unsigned stride0,
                                                unsigned tile_d0,
                                                unsigned tile_d1) {
  unsigned long long ga = (unsigned long long)(size_t)gptr;
  unsigned lds_addr = (unsigned)(size_t)lptr;  // low 32 bits = LDS offset
  v4u g0;
  g0[0] = 1u;                                   // count=1, user descriptor
  g0[1] = lds_addr;                             // lds_addr -> bits[63:32]
  g0[2] = (unsigned)(ga & 0xffffffffu);         // global_addr[31:0]
  g0[3] = (unsigned)((ga >> 32) & 0x01ffffffu)  // global_addr[56:32]
          | (2u << 30);                         // type = 2 ("image")
  v8i g1;
  g1[0] = (int)(1u << 16);                      // wg_mask=0, data_size=1 (2B)
  g1[1] = (int)((tensor_d0 & 0xffffu) << 16);   // tensor_dim0[15:0]
  g1[2] = (int)(((tensor_d0 >> 16) & 0xffffu) | ((tensor_d1 & 0xffffu) << 16));
  g1[3] = (int)(((tensor_d1 >> 16) & 0xffffu) | ((tile_d0 & 0xffffu) << 16));
  g1[4] = (int)(tile_d1 & 0xffffu);             // tile_dim1, tile_dim2=0
  g1[5] = (int)stride0;                         // tensor_dim0_stride[31:0]
  g1[6] = 0;                                    // stride0[47:32], stride1 lo
  g1[7] = 0;                                    // stride1 hi
  v4i g2;
  g2[0] = 0; g2[1] = 0; g2[2] = 0; g2[3] = 0;
  v4i g3 = g2;
#if defined(__clang_major__) && (__clang_major__ >= 23)
  v8i g4;
#pragma unroll
  for (int i = 0; i < 8; ++i) g4[i] = 0;
  __builtin_amdgcn_tensor_load_to_lds(g0, g1, g2, g3, g4, 0);
#else
  __builtin_amdgcn_tensor_load_to_lds(g0, g1, g2, g3, 0);
#endif
}

// A-fragment (16x32 f16): lane row = base + (lane&15).
__device__ __forceinline__ v16h frag_a(const _Float16* p, int lane) {
  const int off = (lane >> 4) * 8;
  v8h lo = *(const v8h*)(p + off);
  v8h hi = *(const v8h*)(p + 16 + off);
  v16h r;
#pragma unroll
  for (int i = 0; i < 8; ++i) { r[i] = lo[i]; r[i + 8] = hi[i]; }
  return r;
}

// B-fragment (32x16 f16): lane col = base + (lane&15); K in table order (same as A).
__device__ __forceinline__ v16h frag_b(const _Float16* p, int lane) {
  const int off = (lane >> 4) * 8;
  v8h lo = *(const v8h*)(p + off);
  v8h hi = *(const v8h*)(p + 16 + off);
  v16h r;
#pragma unroll
  for (int i = 0; i < 8; ++i) { r[i] = lo[i]; r[i + 8] = hi[i]; }
  return r;
}

// ---------------------------------------------------------------- convert
__global__ void cvt_f32_to_f16(const float* __restrict__ in,
                               _Float16* __restrict__ out, int n) {
  int i = blockIdx.x * blockDim.x + threadIdx.x;
  if (i < n) out[i] = (_Float16)in[i];
}

// ---------------------------------------------------------------- QKV GEMM
// C[8192][1536] = A[8192][512] * W[1536][512]^T + bias; scatter q/k/v f16.
// TDM double-buffered tile pipeline: wave 0 DMAs next K-step while all
// 8 waves run WMMAs on the current buffer.
__global__ __launch_bounds__(256)
void qkv_gemm(const _Float16* __restrict__ A, const _Float16* __restrict__ W,
              const float* __restrict__ bias, _Float16* __restrict__ qh,
              _Float16* __restrict__ kh, _Float16* __restrict__ vh) {
  __shared__ _Float16 As[2][128 * 32];
  __shared__ _Float16 Bs[2][128 * 32];
  const int t = threadIdx.x, lane = t & 31, wid = t >> 5;
  const int n0 = blockIdx.x * 128, m0 = blockIdx.y * 128;
  const int wm = (wid & 3) * 32;   // 4 waves along M
  const int wn = (wid >> 2) * 64;  // 2 waves along N

  v8f acc[2][4];
#pragma unroll
  for (int i = 0; i < 2; ++i)
#pragma unroll
    for (int j = 0; j < 4; ++j) acc[i][j] = vzero8f();

  if (wid == 0) {  // prologue: DMA first tiles into buffer 0
    tdm_load_2d_f16(A + (size_t)m0 * 512, &As[0][0], 512, 8192, 512, 32, 128);
    tdm_load_2d_f16(W + (size_t)n0 * 512, &Bs[0][0], 512, 1536, 512, 32, 128);
  }

  for (int k0 = 0; k0 < 512; k0 += 32) {
    const int buf = (k0 >> 5) & 1;
    if (wid == 0) __builtin_amdgcn_s_wait_tensorcnt(0);
    __syncthreads();  // tile[buf] ready for all waves
    if (wid == 0 && k0 + 32 < 512) {
      tdm_load_2d_f16(A + (size_t)m0 * 512 + k0 + 32, &As[buf ^ 1][0],
                      512, 8192, 512, 32, 128);
      tdm_load_2d_f16(W + (size_t)n0 * 512 + k0 + 32, &Bs[buf ^ 1][0],
                      512, 1536, 512, 32, 128);
    }

    v16h a[2];
#pragma unroll
    for (int tm = 0; tm < 2; ++tm)
      a[tm] = frag_a(&As[buf][(wm + tm * 16 + (lane & 15)) * 32], lane);
#pragma unroll
    for (int tn = 0; tn < 4; ++tn) {
      v16h b = frag_b(&Bs[buf][(wn + tn * 16 + (lane & 15)) * 32], lane);
#pragma unroll
      for (int tm = 0; tm < 2; ++tm) acc[tm][tn] = wmma_f16(a[tm], b, acc[tm][tn]);
    }
    __syncthreads();  // compute done before this buffer is refilled
  }

  const int rbase = (lane >> 4) * 8;
#pragma unroll
  for (int tm = 0; tm < 2; ++tm)
#pragma unroll
    for (int tn = 0; tn < 4; ++tn) {
      int n = n0 + wn + tn * 16 + (lane & 15);
      float bv = bias[n];
      int which = n >> 9;  // 0=q 1=k 2=v
      int d = n & 511;
      int h = d >> 6, hd = d & 63;
      _Float16* dst = which == 0 ? qh : (which == 1 ? kh : vh);
#pragma unroll
      for (int r = 0; r < 8; ++r) {
        int m = m0 + wm + tm * 16 + rbase + r;
        int bb = m >> 12, pos = m & 4095;
        dst[((size_t)(bb * 8 + h) * 4096 + pos) * 64 + hd] =
            (_Float16)(acc[tm][tn][r] + bv);
      }
    }
}

// ---------------------------------------------------------------- flash attn
// One block = one (b,h) x 128 Q rows; 8 waves each own 16 rows.
// K tile DMA'd by TDM (double buffered, overlapped with compute);
// V tile manually transposed into LDS; online softmax in f32.
__global__ __launch_bounds__(256)
void flash_attn(const _Float16* __restrict__ qh, const _Float16* __restrict__ kh,
                const _Float16* __restrict__ vh, _Float16* __restrict__ oh) {
  __shared__ _Float16 Ks[2][32 * 64];   // [key][hd], TDM filled
  __shared__ _Float16 Vt[64 * 32];      // [hd][key] (transposed)
  __shared__ _Float16 Ps[8 * 16 * 32];  // per-wave P patch [16][32]
  const int t = threadIdx.x, lane = t & 31, wid = t >> 5;
  const int bh = blockIdx.x;
  const int m0 = blockIdx.y * 128 + wid * 16;
  const _Float16* Q = qh + (size_t)bh * 4096 * 64;
  const _Float16* K = kh + (size_t)bh * 4096 * 64;
  const _Float16* V = vh + (size_t)bh * 4096 * 64;
  _Float16* Pw = &Ps[wid * 16 * 32];

  const int qrow = m0 + (lane & 15);
  v16h qa0 = frag_a(Q + (size_t)qrow * 64 + 0, lane);
  v16h qa1 = frag_a(Q + (size_t)qrow * 64 + 32, lane);

  float mrow[8], lrow[8];
  v8f acc[4];
#pragma unroll
  for (int r = 0; r < 8; ++r) { mrow[r] = -1e30f; lrow[r] = 0.0f; }
#pragma unroll
  for (int n = 0; n < 4; ++n) acc[n] = vzero8f();

  const float SC = 0.125f;  // 1/sqrt(64)

  if (wid == 0)  // prologue: first K tile
    tdm_load_2d_f16(K, &Ks[0][0], 64, 4096, 64, 64, 32);

  for (int j = 0; j < 4096; j += 32) {
    const int buf = (j >> 5) & 1;
    {  // cooperative V tile load + transpose (safe: prior reads fenced by
       // the trailing barrier of the previous iteration)
      int idx = t * 8;
      int row = idx >> 6, col = idx & 63;
      v8h vv = *(const v8h*)&V[(size_t)(j + row) * 64 + col];
#pragma unroll
      for (int i = 0; i < 8; ++i) Vt[(col + i) * 32 + row] = vv[i];
    }
    if (wid == 0) __builtin_amdgcn_s_wait_tensorcnt(0);
    __syncthreads();  // K[buf] (TDM) + Vt visible to all waves
    if (wid == 0 && j + 32 < 4096)
      tdm_load_2d_f16(K + (size_t)(j + 32) * 64, &Ks[buf ^ 1][0],
                      64, 4096, 64, 64, 32);

    // S = Q * K^T for two 16-key sub-tiles (Hd=64 -> 2 WMMA k-steps each)
    v8f s[2];
#pragma unroll
    for (int kt = 0; kt < 2; ++kt) {
      v16h b0 = frag_b(&Ks[buf][(kt * 16 + (lane & 15)) * 64 + 0], lane);
      v16h b1 = frag_b(&Ks[buf][(kt * 16 + (lane & 15)) * 64 + 32], lane);
      v8f z = vzero8f();
      z = wmma_f16(qa0, b0, z);
      z = wmma_f16(qa1, b1, z);
      s[kt] = z;
    }

    // online softmax (row reductions across the 16 lanes of each half-wave)
    float fac[8];
#pragma unroll
    for (int r = 0; r < 8; ++r) {
      float v0 = s[0][r] * SC, v1 = s[1][r] * SC;
      float mx = fmaxf(v0, v1);
      mx = fmaxf(mx, __shfl_xor(mx, 1, 32));
      mx = fmaxf(mx, __shfl_xor(mx, 2, 32));
      mx = fmaxf(mx, __shfl_xor(mx, 4, 32));
      mx = fmaxf(mx, __shfl_xor(mx, 8, 32));
      float nm = fmaxf(mrow[r], mx);
      float f = __expf(mrow[r] - nm);
      float p0 = __expf(v0 - nm), p1 = __expf(v1 - nm);
      s[0][r] = p0; s[1][r] = p1;
      float rs = p0 + p1;
      rs += __shfl_xor(rs, 1, 32);
      rs += __shfl_xor(rs, 2, 32);
      rs += __shfl_xor(rs, 4, 32);
      rs += __shfl_xor(rs, 8, 32);
      lrow[r] = lrow[r] * f + rs;
      mrow[r] = nm;
      fac[r] = f;
    }
#pragma unroll
    for (int n = 0; n < 4; ++n)
#pragma unroll
      for (int r = 0; r < 8; ++r) acc[n][r] *= fac[r];

    // C-layout P -> per-wave LDS patch -> A-layout fragment (intra-wave)
    {
      int prow = (lane >> 4) * 8, col = lane & 15;
#pragma unroll
      for (int r = 0; r < 8; ++r) {
        Pw[(prow + r) * 32 + col] = (_Float16)s[0][r];
        Pw[(prow + r) * 32 + 16 + col] = (_Float16)s[1][r];
      }
    }
    asm volatile("s_wait_dscnt 0" ::: "memory");  // DS in-order: P visible
    v16h pa = frag_a(&Pw[(lane & 15) * 32], lane);
#pragma unroll
    for (int n = 0; n < 4; ++n) {
      v16h vb = frag_b(&Vt[(n * 16 + (lane & 15)) * 32], lane);
      acc[n] = wmma_f16(pa, vb, acc[n]);
    }
    __syncthreads();  // Vt/Ps reads done before next iteration's refill
  }

  // normalize + store O as f16 [b][token][h*64+hd] for the out projection
  const int rbase = (lane >> 4) * 8;
  float inv[8];
#pragma unroll
  for (int r = 0; r < 8; ++r) inv[r] = 1.0f / lrow[r];
  const int b = bh >> 3, h = bh & 7;
#pragma unroll
  for (int n = 0; n < 4; ++n) {
    int hd = n * 16 + (lane & 15);
#pragma unroll
    for (int r = 0; r < 8; ++r) {
      int token = m0 + rbase + r;
      oh[((size_t)(b * 4096 + token) * 512) + h * 64 + hd] =
          (_Float16)(acc[n][r] * inv[r]);
    }
  }
}

// ---------------------------------------------------------------- out GEMM
// out[8192][512] = O[8192][512] * Wo[512][512]^T + bias (f32 result)
__global__ __launch_bounds__(256)
void out_gemm(const _Float16* __restrict__ A, const _Float16* __restrict__ W,
              const float* __restrict__ bias, float* __restrict__ out) {
  __shared__ _Float16 As[2][128 * 32];
  __shared__ _Float16 Bs[2][128 * 32];
  const int t = threadIdx.x, lane = t & 31, wid = t >> 5;
  const int n0 = blockIdx.x * 128, m0 = blockIdx.y * 128;
  const int wm = (wid & 3) * 32;
  const int wn = (wid >> 2) * 64;

  v8f acc[2][4];
#pragma unroll
  for (int i = 0; i < 2; ++i)
#pragma unroll
    for (int j = 0; j < 4; ++j) acc[i][j] = vzero8f();

  if (wid == 0) {
    tdm_load_2d_f16(A + (size_t)m0 * 512, &As[0][0], 512, 8192, 512, 32, 128);
    tdm_load_2d_f16(W + (size_t)n0 * 512, &Bs[0][0], 512, 512, 512, 32, 128);
  }

  for (int k0 = 0; k0 < 512; k0 += 32) {
    const int buf = (k0 >> 5) & 1;
    if (wid == 0) __builtin_amdgcn_s_wait_tensorcnt(0);
    __syncthreads();
    if (wid == 0 && k0 + 32 < 512) {
      tdm_load_2d_f16(A + (size_t)m0 * 512 + k0 + 32, &As[buf ^ 1][0],
                      512, 8192, 512, 32, 128);
      tdm_load_2d_f16(W + (size_t)n0 * 512 + k0 + 32, &Bs[buf ^ 1][0],
                      512, 512, 512, 32, 128);
    }

    v16h a[2];
#pragma unroll
    for (int tm = 0; tm < 2; ++tm)
      a[tm] = frag_a(&As[buf][(wm + tm * 16 + (lane & 15)) * 32], lane);
#pragma unroll
    for (int tn = 0; tn < 4; ++tn) {
      v16h b = frag_b(&Bs[buf][(wn + tn * 16 + (lane & 15)) * 32], lane);
#pragma unroll
      for (int tm = 0; tm < 2; ++tm) acc[tm][tn] = wmma_f16(a[tm], b, acc[tm][tn]);
    }
    __syncthreads();
  }

  const int rbase = (lane >> 4) * 8;
#pragma unroll
  for (int tm = 0; tm < 2; ++tm)
#pragma unroll
    for (int tn = 0; tn < 4; ++tn) {
      int n = n0 + wn + tn * 16 + (lane & 15);
      float bv = bias[n];
#pragma unroll
      for (int r = 0; r < 8; ++r) {
        int m = m0 + wm + tm * 16 + rbase + r;
        out[(size_t)m * 512 + n] = acc[tm][tn][r] + bv;
      }
    }
}

// ---------------------------------------------------------------- launcher
extern "C" void kernel_launch(void* const* d_in, const int* in_sizes, int n_in,
                              void* d_out, int out_size, void* d_ws,
                              size_t ws_size, hipStream_t stream) {
  (void)in_sizes; (void)n_in; (void)out_size; (void)ws_size;
  const float* x = (const float*)d_in[0];
  const float* qkv_w = (const float*)d_in[1];
  const float* qkv_b = (const float*)d_in[2];
  const float* out_w = (const float*)d_in[3];
  const float* out_b = (const float*)d_in[4];
  float* out = (float*)d_out;

  _Float16* ws = (_Float16*)d_ws;
  size_t off = 0;
  _Float16* xh = ws + off;    off += (size_t)8192 * 512;   // x f16
  _Float16* wqkvh = ws + off; off += (size_t)1536 * 512;   // qkv_w f16
  _Float16* wouth = ws + off; off += (size_t)512 * 512;    // out_w f16
  _Float16* qh = ws + off;    off += (size_t)16 * 4096 * 64;
  _Float16* kh = ws + off;    off += (size_t)16 * 4096 * 64;
  _Float16* vh = ws + off;    off += (size_t)16 * 4096 * 64;
  _Float16* oh = ws + off;    off += (size_t)8192 * 512;   // attn output f16

  cvt_f32_to_f16<<<(8192 * 512 + 255) / 256, 256, 0, stream>>>(x, xh, 8192 * 512);
  cvt_f32_to_f16<<<(1536 * 512 + 255) / 256, 256, 0, stream>>>(qkv_w, wqkvh, 1536 * 512);
  cvt_f32_to_f16<<<(512 * 512 + 255) / 256, 256, 0, stream>>>(out_w, wouth, 512 * 512);

  qkv_gemm<<<dim3(12, 64), 256, 0, stream>>>(xh, wqkvh, qkv_b, qh, kh, vh);
  flash_attn<<<dim3(16, 32), 256, 0, stream>>>(qh, kh, vh, oh);
  out_gemm<<<dim3(4, 64), 256, 0, stream>>>(oh, wouth, out_b, out);
}
